// LinearSelfAttention_360777253379
// MI455X (gfx1250) — matrix operand order repaired and checked by reading, not compiled, not run on hardware
//
#include <hip/hip_runtime.h>
#include <hip/hip_bf16.h>

typedef __bf16 bf16_t;
typedef bf16_t v16bf __attribute__((ext_vector_type(16)));
typedef float  v8f   __attribute__((ext_vector_type(8)));

#define CDIM 384
#define NTOT 4096
#define PDIM 4
#define BDIM 16
// packed fragment array: [mt(24)][kk(12)][lane(32)][elem(16)] bf16
#define FRAG_ELEMS (24 * 12 * 32 * 16) // 147456

// ---------------------------------------------------------------------------
// Kernel 0: pack Wv (rows 385..768 of w_qkv) and Wout into WMMA A-fragment
// order, split into bf16 hi/lo pairs (Markidis split for ~fp32 accuracy).
// A 16x32 order used here (contiguous k; B reads match): lane<16 -> M=lane, elem i => K=i;
//                        lane>=16 -> M=lane-16, elem i => K=i+16.
// ---------------------------------------------------------------------------
__global__ __launch_bounds__(256) void pack_weights_kernel(
    const float* __restrict__ w_qkv, const float* __restrict__ w_out,
    bf16_t* __restrict__ wvhi, bf16_t* __restrict__ wvlo,
    bf16_t* __restrict__ wohi, bf16_t* __restrict__ wolo)
{
    int e    = blockIdx.x * 256 + threadIdx.x;     // < 147456 (grid sized exactly)
    int i    = e & 15;
    int lane = (e >> 4) & 31;
    int kk   = (e >> 9) % 12;
    int mt   = e / 6144;
    int m = mt * 16 + (lane & 15);
    int k = kk * 32 + ((lane < 16) ? i : i + 16);
    float wv = w_qkv[(size_t)(385 + m) * CDIM + k];  // value projection rows
    bf16_t h = (bf16_t)wv;
    wvhi[e] = h;
    wvlo[e] = (bf16_t)(wv - (float)h);

    float wo = w_out[(size_t)m * CDIM + k];
    bf16_t h2 = (bf16_t)wo;
    wohi[e] = h2;
    wolo[e] = (bf16_t)(wo - (float)h2);
}

// ---------------------------------------------------------------------------
// Kernel 1: q[n] = dot(w_qkv[0,:], x[b,:,p,n]) + b_qkv[0]; softmax over n.
// One block per (b,p). Each thread owns 16 of the 4096 positions.
// ---------------------------------------------------------------------------
__global__ __launch_bounds__(256) void q_softmax_kernel(
    const float* __restrict__ x, const float* __restrict__ w_qkv,
    const float* __restrict__ b_qkv, float* __restrict__ scores)
{
    __shared__ float red[256];
    const int tid = threadIdx.x;
    const int bp = blockIdx.x;
    const int b = bp >> 2, p = bp & 3;

    float acc[16];
    const float bq = b_qkv[0];
#pragma unroll
    for (int j = 0; j < 16; ++j) acc[j] = bq;

    for (int c = 0; c < CDIM; ++c) {
        const float w = w_qkv[c];
        const float* xr = x + (((size_t)(b * CDIM + c) * PDIM + p) * NTOT);
#pragma unroll
        for (int j = 0; j < 16; ++j) acc[j] += w * xr[tid + j * 256];
    }

    // block max
    float m = acc[0];
#pragma unroll
    for (int j = 1; j < 16; ++j) m = fmaxf(m, acc[j]);
    red[tid] = m; __syncthreads();
    for (int o = 128; o > 0; o >>= 1) {
        if (tid < o) red[tid] = fmaxf(red[tid], red[tid + o]);
        __syncthreads();
    }
    m = red[0]; __syncthreads();

    // exp + block sum
    float s = 0.f;
#pragma unroll
    for (int j = 0; j < 16; ++j) { acc[j] = __expf(acc[j] - m); s += acc[j]; }
    red[tid] = s; __syncthreads();
    for (int o = 128; o > 0; o >>= 1) {
        if (tid < o) red[tid] += red[tid + o];
        __syncthreads();
    }
    const float inv = 1.0f / red[0];

    float* sr = scores + (size_t)bp * NTOT;
#pragma unroll
    for (int j = 0; j < 16; ++j) sr[tid + j * 256] = acc[j] * inv;
}

// ---------------------------------------------------------------------------
// Kernel 2: xs[bp,c] = sum_n x[b,c,p,n] * scores[bp,n]. One block per (b,p).
// Wave per channel, lanes reduce over n.
// ---------------------------------------------------------------------------
__global__ __launch_bounds__(256) void xs_kernel(
    const float* __restrict__ x, const float* __restrict__ scores,
    float* __restrict__ xs)
{
    __shared__ float sc[NTOT];
    const int tid = threadIdx.x;
    const int bp = blockIdx.x;
    const int b = bp >> 2, p = bp & 3;
    const float* sr = scores + (size_t)bp * NTOT;
#pragma unroll
    for (int j = 0; j < 16; ++j) sc[tid + j * 256] = sr[tid + j * 256];
    __syncthreads();

    const int wave = tid >> 5, lane = tid & 31;
    for (int c = wave; c < CDIM; c += 8) {
        const float* xr = x + (((size_t)(b * CDIM + c) * PDIM + p) * NTOT);
        float a = 0.f;
#pragma unroll 4
        for (int n = lane; n < NTOT; n += 32) a += xr[n] * sc[n];
#pragma unroll
        for (int o = 16; o > 0; o >>= 1) a += __shfl_xor(a, o, 32);
        if (lane == 0) xs[(size_t)bp * CDIM + c] = a;
    }
}

// ---------------------------------------------------------------------------
// Kernel 3: cv[bp,c] = sum_{c'} w_k[c,c'] * xs[bp,c'] + b_k[c]
// (softmax sums to 1, so the key bias passes through exactly). Tiny GEMV.
// ---------------------------------------------------------------------------
__global__ __launch_bounds__(256) void cv_kernel(
    const float* __restrict__ w_qkv, const float* __restrict__ b_qkv,
    const float* __restrict__ xs, float* __restrict__ cv)
{
    __shared__ float xsl[CDIM];
    const int tid = threadIdx.x;
    const int bp = blockIdx.x;
    for (int idx = tid; idx < CDIM; idx += 256) xsl[idx] = xs[(size_t)bp * CDIM + idx];
    __syncthreads();

    for (int c = tid; c < CDIM; c += 256) {
        const float* wr = w_qkv + (size_t)(1 + c) * CDIM;  // key rows 1..384
        float a = b_qkv[1 + c];
#pragma unroll 4
        for (int k = 0; k < CDIM; ++k) a += wr[k] * xsl[k];
        cv[(size_t)bp * CDIM + c] = a;
    }
}

// ---------------------------------------------------------------------------
// Kernel 4 (the heavy one): per (b,p, 64-wide n tile):
//   V = Wv * X          (384 x 64, split-bf16 WMMA, fp32 accum)
//   U = relu(V + bv) * cv[c]   -> LDS in B-fragment layout (bf16 hi/lo)
//   O = Wout * U + bo   (384 x 64, split-bf16 WMMA)
// 8 waves; wave w owns rows [48w, 48w+48) = 3 M-tiles x 4 N-tiles.
// ---------------------------------------------------------------------------
__global__ __launch_bounds__(256, 1) void fused_vo_kernel(
    const float* __restrict__ x,
    const bf16_t* __restrict__ wvhi, const bf16_t* __restrict__ wvlo,
    const bf16_t* __restrict__ wohi, const bf16_t* __restrict__ wolo,
    const float* __restrict__ b_qkv, const float* __restrict__ b_out,
    const float* __restrict__ cv, float* __restrict__ out)
{
    __shared__ __align__(32) bf16_t Xhi[64][48];    // [n][k], padded stride
    __shared__ __align__(32) bf16_t Xlo[64][48];
    __shared__ __align__(32) bf16_t Uhi[64][400];   // [n][c], padded stride
    __shared__ __align__(32) bf16_t Ulo[64][400];
    __shared__ float cvL[CDIM], bvL[CDIM], boL[CDIM];

    const int tid  = threadIdx.x;
    const int wave = tid >> 5;
    const int lane = tid & 31;
    const int bp   = blockIdx.y;
    const int b = bp >> 2, p = bp & 3;
    const int n0 = blockIdx.x * 64;

    for (int idx = tid; idx < CDIM; idx += 256) {
        cvL[idx] = cv[(size_t)bp * CDIM + idx];
        bvL[idx] = b_qkv[385 + idx];   // value bias
        boL[idx] = b_out[idx];
    }

    const int kloc = tid >> 3;          // 0..31 : k row to fetch
    const int ngrp = (tid & 7) * 8;     // 0..56 : n group of 8
    const int brow = lane & 15;         // B-fragment n row within tile
    const int bko  = (lane < 16) ? 0 : 16;

    v8f acc[3][4] = {};

    // ---- Stage A: V = Wv * X, K = 384 in steps of 32 ----
    for (int kk = 0; kk < 12; ++kk) {
        __syncthreads();
        {   // cooperative load + fp32->bf16 hi/lo split + transpose into LDS
            const float* xp = x + (((size_t)(b * CDIM + kk * 32 + kloc) * PDIM + p) * NTOT
                                   + n0 + ngrp);
            float4 v0 = ((const float4*)xp)[0];
            float4 v1 = ((const float4*)xp)[1];
            float vv[8] = {v0.x, v0.y, v0.z, v0.w, v1.x, v1.y, v1.z, v1.w};
#pragma unroll
            for (int j = 0; j < 8; ++j) {
                bf16_t h = (bf16_t)vv[j];
                Xhi[ngrp + j][kloc] = h;
                Xlo[ngrp + j][kloc] = (bf16_t)(vv[j] - (float)h);
            }
        }
        __syncthreads();

        v16bf ahi[3], alo[3];
#pragma unroll
        for (int i = 0; i < 3; ++i) {
            size_t fo = (((size_t)(wave * 3 + i) * 12 + kk) * 32 + lane) * 16;
            ahi[i] = *(const v16bf*)(wvhi + fo);
            alo[i] = *(const v16bf*)(wvlo + fo);
        }
#pragma unroll
        for (int j = 0; j < 4; ++j) {
            v16bf bhi = *(const v16bf*)&Xhi[j * 16 + brow][bko];
            v16bf blo = *(const v16bf*)&Xlo[j * 16 + brow][bko];
#pragma unroll
            for (int i = 0; i < 3; ++i) {
                acc[i][j] = __builtin_amdgcn_wmma_f32_16x16x32_bf16(
                    false, ahi[i], false, bhi, (short)0, acc[i][j], false, false);
                acc[i][j] = __builtin_amdgcn_wmma_f32_16x16x32_bf16(
                    false, ahi[i], false, blo, (short)0, acc[i][j], false, false);
                acc[i][j] = __builtin_amdgcn_wmma_f32_16x16x32_bf16(
                    false, alo[i], false, bhi, (short)0, acc[i][j], false, false);
            }
        }
    }

    // ---- U = relu(V + bv) * cv, written to LDS in B-fragment layout ----
#pragma unroll
    for (int i = 0; i < 3; ++i) {
#pragma unroll
        for (int j = 0; j < 4; ++j) {
#pragma unroll
            for (int r = 0; r < 8; ++r) {
                int c  = wave * 48 + i * 16 + r + ((lane & 16) ? 8 : 0);
                int nl = j * 16 + brow;
                float v = acc[i][j][r] + bvL[c];
                v = fmaxf(v, 0.0f) * cvL[c];
                bf16_t h = (bf16_t)v;
                Uhi[nl][c] = h;
                Ulo[nl][c] = (bf16_t)(v - (float)h);
            }
        }
    }
    __syncthreads();

    // ---- Stage B: O = Wout * U ----
    v8f acc2[3][4] = {};
    for (int kk = 0; kk < 12; ++kk) {
        v16bf ahi[3], alo[3];
#pragma unroll
        for (int i = 0; i < 3; ++i) {
            size_t fo = (((size_t)(wave * 3 + i) * 12 + kk) * 32 + lane) * 16;
            ahi[i] = *(const v16bf*)(wohi + fo);
            alo[i] = *(const v16bf*)(wolo + fo);
        }
#pragma unroll
        for (int j = 0; j < 4; ++j) {
            v16bf bhi = *(const v16bf*)&Uhi[j * 16 + brow][kk * 32 + bko];
            v16bf blo = *(const v16bf*)&Ulo[j * 16 + brow][kk * 32 + bko];
#pragma unroll
            for (int i = 0; i < 3; ++i) {
                acc2[i][j] = __builtin_amdgcn_wmma_f32_16x16x32_bf16(
                    false, ahi[i], false, bhi, (short)0, acc2[i][j], false, false);
                acc2[i][j] = __builtin_amdgcn_wmma_f32_16x16x32_bf16(
                    false, ahi[i], false, blo, (short)0, acc2[i][j], false, false);
                acc2[i][j] = __builtin_amdgcn_wmma_f32_16x16x32_bf16(
                    false, alo[i], false, bhi, (short)0, acc2[i][j], false, false);
            }
        }
    }

    // ---- epilogue: out = O + b_out ----
#pragma unroll
    for (int i = 0; i < 3; ++i) {
#pragma unroll
        for (int j = 0; j < 4; ++j) {
#pragma unroll
            for (int r = 0; r < 8; ++r) {
                int c  = wave * 48 + i * 16 + r + ((lane & 16) ? 8 : 0);
                int nl = j * 16 + brow;
                out[(((size_t)(b * CDIM + c) * PDIM + p) * NTOT) + n0 + nl] =
                    acc2[i][j][r] + boL[c];
            }
        }
    }
}

// ---------------------------------------------------------------------------
extern "C" void kernel_launch(void* const* d_in, const int* in_sizes, int n_in,
                              void* d_out, int out_size, void* d_ws, size_t ws_size,
                              hipStream_t stream) {
    const float* x     = (const float*)d_in[0];
    const float* w_qkv = (const float*)d_in[1];
    const float* b_qkv = (const float*)d_in[2];
    const float* w_out = (const float*)d_in[3];
    const float* b_out = (const float*)d_in[4];
    float* out = (float*)d_out;

    // workspace layout (~2.4 MB total)
    char* ws = (char*)d_ws;
    float*  scores = (float*)ws;                          // 64*4096 f32 = 1 MB
    bf16_t* wvhi   = (bf16_t*)(ws + (1u << 20));          // 147456 bf16 each
    bf16_t* wvlo   = wvhi + FRAG_ELEMS;
    bf16_t* wohi   = wvlo + FRAG_ELEMS;
    bf16_t* wolo   = wohi + FRAG_ELEMS;
    float*  xs     = (float*)(ws + (1u << 20) + 4u * FRAG_ELEMS * sizeof(bf16_t));
    float*  cv     = xs + 64 * CDIM;

    pack_weights_kernel<<<FRAG_ELEMS / 256, 256, 0, stream>>>(w_qkv, w_out,
                                                              wvhi, wvlo, wohi, wolo);
    q_softmax_kernel<<<BDIM * PDIM, 256, 0, stream>>>(x, w_qkv, b_qkv, scores);
    xs_kernel<<<BDIM * PDIM, 256, 0, stream>>>(x, scores, xs);
    cv_kernel<<<BDIM * PDIM, 256, 0, stream>>>(w_qkv, b_qkv, xs, cv);
    fused_vo_kernel<<<dim3(NTOT / 64, BDIM * PDIM), 256, 0, stream>>>(
        x, wvhi, wvlo, wohi, wolo, b_qkv, b_out, cv, out);
}
